// BiLSTM_CRF_70222715289995
// MI455X (gfx1250) — matrix operand order repaired and checked by reading, not compiled, not run on hardware
//
#include <hip/hip_runtime.h>
#include <hip/hip_bf16.h>
#include <cstdint>
#include <cstddef>

// ---------------------------------------------------------------------------
// Problem constants (match reference)
// ---------------------------------------------------------------------------
#define B_SZ 512
#define T_SZ 256
#define V_SZ 128
#define E_SZ 64
#define H_SZ 64   // per-direction hidden
// gates = 4*H = 256

typedef __attribute__((ext_vector_type(16))) _Float16 v16h;
typedef __attribute__((ext_vector_type(8)))  _Float16 v8h;
typedef __attribute__((ext_vector_type(8)))  float    v8f;

// ---------------------------------------------------------------------------
// WMMA fragment loaders (CDNA5 wave32 layouts, 16-bit operands).
// A (16x32 MxK): lane l holds row (l&15); K-halves 0/8 select v0-3 base,
//   +16 for v4-7  -> two 16-byte loads per fragment.
// B (32x16 KxN): lane l holds col (l&15); same K order as A (K-halves 0/8,
//   +16 for v4-7) -> two 16-byte loads per fragment.
// Works for LDS or global pointers (K-contiguous row-major storage).
// ---------------------------------------------------------------------------
static __device__ __forceinline__ v16h load_frag_a(const _Float16* A, int lda,
                                                   int rowbase, int kb, int lane) {
  int r  = rowbase + (lane & 15);
  int kh = kb + ((lane >> 4) << 3);
  const _Float16* p = A + r * lda + kh;
  v8h lo = *(const v8h*)(p);
  v8h hi = *(const v8h*)(p + 16);
  v16h o;
#pragma unroll
  for (int i = 0; i < 8; ++i) { o[i] = lo[i]; o[i + 8] = hi[i]; }
  return o;
}

static __device__ __forceinline__ v16h load_frag_b(const _Float16* Wm, int ldb,
                                                   int ncol, int kb, int lane) {
  int n  = ncol + (lane & 15);
  int kh = kb + ((lane >> 4) << 3);
  const _Float16* p = Wm + n * ldb + kh;
  v8h lo = *(const v8h*)(p);
  v8h hi = *(const v8h*)(p + 16);
  v16h o;
#pragma unroll
  for (int i = 0; i < 8; ++i) { o[i] = lo[i]; o[i + 8] = hi[i]; }
  return o;
}

static __device__ __forceinline__ float sigf(float x) {
  return 1.0f / (1.0f + __expf(-x));
}

// Fast tanh built on v_exp_f32 (avoids ocml tanh call chain).
static __device__ __forceinline__ float fast_tanh(float x) {
  float ax = fabsf(x);
  float e  = __expf(-2.0f * ax);
  float t  = 1.0f - 2.0f * e / (1.0f + e);
  return copysignf(t, x);
}

// ---------------------------------------------------------------------------
// Pack weights: fp32 -> f16, [W_ih | W_hh] concatenated along K, row-major
// (rows = 256 gate outputs, K contiguous). Combined bias = b_ih + b_hh.
// ---------------------------------------------------------------------------
__global__ void pack_weights_kernel(
    const float* __restrict__ wih0,  const float* __restrict__ whh0,
    const float* __restrict__ bih0,  const float* __restrict__ bhh0,
    const float* __restrict__ wih0r, const float* __restrict__ whh0r,
    const float* __restrict__ bih0r, const float* __restrict__ bhh0r,
    const float* __restrict__ wih1,  const float* __restrict__ whh1,
    const float* __restrict__ bih1,  const float* __restrict__ bhh1,
    const float* __restrict__ wih1r, const float* __restrict__ whh1r,
    const float* __restrict__ bih1r, const float* __restrict__ bhh1r,
    const float* __restrict__ linw,
    _Float16* __restrict__ Wl0f, _Float16* __restrict__ Wl0r,
    _Float16* __restrict__ Wl1f, _Float16* __restrict__ Wl1r,
    float* __restrict__ biasAll, _Float16* __restrict__ lwh) {
  const int stride = gridDim.x * blockDim.x;
  for (int idx = blockIdx.x * blockDim.x + threadIdx.x; idx < 256 * 192; idx += stride) {
    if (idx < 256 * 128) {  // layer0: K = 64 + 64
      int n = idx >> 7, k = idx & 127;
      float vf = (k < 64) ? wih0 [n * 64 + k] : whh0 [n * 64 + (k - 64)];
      float vr = (k < 64) ? wih0r[n * 64 + k] : whh0r[n * 64 + (k - 64)];
      Wl0f[idx] = (_Float16)vf;
      Wl0r[idx] = (_Float16)vr;
    }
    {  // layer1: K = 128 + 64
      int n = idx / 192, k = idx % 192;
      float vf = (k < 128) ? wih1 [n * 128 + k] : whh1 [n * 64 + (k - 128)];
      float vr = (k < 128) ? wih1r[n * 128 + k] : whh1r[n * 64 + (k - 128)];
      Wl1f[idx] = (_Float16)vf;
      Wl1r[idx] = (_Float16)vr;
    }
    if (idx < V_SZ * V_SZ) lwh[idx] = (_Float16)linw[idx];
    if (idx < 256) {
      biasAll[idx]       = bih0 [idx] + bhh0 [idx];
      biasAll[256 + idx] = bih0r[idx] + bhh0r[idx];
      biasAll[512 + idx] = bih1 [idx] + bhh1 [idx];
      biasAll[768 + idx] = bih1r[idx] + bhh1r[idx];
    }
  }
}

// ---------------------------------------------------------------------------
// Embedding gather -> f16 (B,T,E)
// ---------------------------------------------------------------------------
__global__ void embed_kernel(const int* __restrict__ x, const float* __restrict__ emb,
                             _Float16* __restrict__ out) {
  const size_t total  = (size_t)B_SZ * T_SZ * E_SZ;
  const size_t stride = (size_t)gridDim.x * blockDim.x;
  for (size_t idx = (size_t)blockIdx.x * blockDim.x + threadIdx.x; idx < total; idx += stride) {
    size_t bt = idx >> 6;
    int e = (int)(idx & 63);
    out[idx] = (_Float16)emb[(size_t)x[bt] * E_SZ + e];
  }
}

// ---------------------------------------------------------------------------
// Fused LSTM direction kernel.
//   D = input feature dim (64 layer0, 128 layer1); K = D + H.
//   grid = (B/16, 2 directions), 128 threads = 4 waves  -> 64 blocks/layer,
//   spreading the latency-bound recurrence across more WGPs.
//   LDS: W (256 x K, f16), A-tile (16 x K, f16: cols [0,D)=x_t, [D,K)=h),
//        combined bias (256 f32).
//   Wave w owns the 16-wide slice [16w,16w+16) of each of the 4 gates, so the
//   cell update is wave-local (i/f/g/o for a (row,hidden) pair in one lane).
//   x_t staging uses gfx1250 async LDS loads (ASYNCcnt) via inline asm.
// ---------------------------------------------------------------------------
template <int D>
__global__ __launch_bounds__(128) void lstm_dir_kernel(
    const _Float16* __restrict__ in,   // (B,T,D) f16
    _Float16* __restrict__ out,        // (B,T,2H) f16, writes [dir*H, dir*H+H)
    const _Float16* __restrict__ Wfwd, // (256,K) f16
    const _Float16* __restrict__ Wrev,
    const float* __restrict__ biasFwd, // 256
    const float* __restrict__ biasRev) {
  constexpr int K  = D + H_SZ;
  constexpr int BM = 16;

  extern __shared__ char smem[];
  _Float16* Ws = (_Float16*)smem;        // 256*K halves
  _Float16* As = Ws + 256 * K;           // BM*K halves
  float*    bs = (float*)(As + BM * K);  // 256 floats
  const unsigned aoff = (unsigned)((char*)As - smem);  // LDS byte offset of A

  const int dir   = blockIdx.y;
  const int bbase = blockIdx.x * BM;
  const _Float16* W    = dir ? Wrev : Wfwd;
  const float*    bias = dir ? biasRev : biasFwd;

  const int tid  = threadIdx.x;
  const int lane = tid & 31;
  const int wv   = tid >> 5;             // 0..3

  // Stage weights into LDS (resident across all T steps).
  for (int idx = tid * 8; idx < 256 * K; idx += 128 * 8)
    *(v8h*)(Ws + idx) = *(const v8h*)(W + idx);
  bs[tid]       = bias[tid];
  bs[tid + 128] = bias[tid + 128];
  // h(t=-1) = 0
  for (int idx = tid; idx < BM * H_SZ; idx += 128) {
    int r = idx >> 6, c = idx & 63;
    As[r * K + D + c] = (_Float16)0.0f;
  }
  __syncthreads();

  const int cbase = wv * 16;             // 16-wide slice inside each 64-wide gate
  const int mrow  = (lane >> 4) << 3;    // row base for this lane's acc elems
  const int ncol  = cbase + (lane & 15); // hidden index this lane owns

  float bgl[4];
#pragma unroll
  for (int g = 0; g < 4; ++g) bgl[g] = bs[g * 64 + ncol];

  float creg[8];
#pragma unroll
  for (int v = 0; v < 8; ++v) creg[v] = 0.0f;

  constexpr int CHUNKS = BM * (D / 8);   // 16-byte chunks per x_t tile

  for (int step = 0; step < T_SZ; ++step) {
    const int t = dir ? (T_SZ - 1 - step) : step;

    // Prefetch next step's x rows into cache (global_prefetch_b8).
    if (step + 1 < T_SZ && tid < BM) {
      const int tn = dir ? (T_SZ - 2 - step) : (step + 1);
      __builtin_prefetch(in + ((size_t)(bbase + tid) * T_SZ + tn) * D, 0, 1);
    }

    // Stage x_t tile into A[:, 0:D) with async global->LDS copies (ASYNCcnt).
#pragma unroll
    for (int idx = tid; idx < CHUNKS; idx += 128) {
      const int r  = idx / (D / 8);
      const int c8 = idx % (D / 8);
      const unsigned ldsoff = aoff + (unsigned)(r * K * 2 + c8 * 16);
      const unsigned long long gaddr =
          (unsigned long long)(const void*)(in + ((size_t)(bbase + r) * T_SZ + t) * D + c8 * 8);
      asm volatile("global_load_async_to_lds_b128 %0, %1, off"
                   :: "v"(ldsoff), "v"(gaddr)
                   : "memory");
    }
    asm volatile("s_wait_asynccnt 0x0" ::: "memory");
    __syncthreads();

    // g = [x_t || h] @ [W_ih ; W_hh]^T  via v_wmma_f32_16x16x32_f16
    v8f acc[4];
#pragma unroll
    for (int g = 0; g < 4; ++g)
#pragma unroll
      for (int v = 0; v < 8; ++v) acc[g][v] = 0.0f;

#pragma unroll
    for (int kb = 0; kb < K; kb += 32) {
      v16h a = load_frag_a(As, K, 0, kb, lane);
#pragma unroll
      for (int g = 0; g < 4; ++g) {
        v16h b = load_frag_b(Ws, K, g * 64 + cbase, kb, lane);
        acc[g] = __builtin_amdgcn_wmma_f32_16x16x32_f16(
            false, a, false, b, (short)0, acc[g], false, false);
      }
    }
    __syncthreads();  // all A reads done before h is overwritten

    // Cell update (fp32), write h to LDS (for t+1) and to global output.
#pragma unroll
    for (int v = 0; v < 8; ++v) {
      float gi = acc[0][v] + bgl[0];
      float gf = acc[1][v] + bgl[1];
      float gg = acc[2][v] + bgl[2];
      float go = acc[3][v] + bgl[3];
      float cc = sigf(gf) * creg[v] + sigf(gi) * fast_tanh(gg);
      float hh = sigf(go) * fast_tanh(cc);
      creg[v] = cc;
      _Float16 hph = (_Float16)hh;
      int r = mrow + v;
      As[r * K + D + ncol] = hph;
      out[((size_t)(bbase + r) * T_SZ + t) * (2 * H_SZ) + dir * H_SZ + ncol] = hph;
    }
    __syncthreads();
  }
}

// ---------------------------------------------------------------------------
// Emissions GEMM: (B*T,128) x (128,128)^T + bias, f16 operands / f32 out.
// Fragments loaded straight from global (layouts already K-contiguous).
// ---------------------------------------------------------------------------
__global__ __launch_bounds__(256) void emissions_kernel(
    const _Float16* __restrict__ h2,   // (B*T, 128) f16
    const _Float16* __restrict__ lwh,  // (V=128, 128) f16 row-major
    const float* __restrict__ lb,      // (V)
    float* __restrict__ em) {          // (B*T, V) f32
  const int lane = threadIdx.x & 31;
  const int wv   = threadIdx.x >> 5;
  const size_t rowblk = (size_t)blockIdx.x * 64;
  const int rbase  = (wv & 3) * 16;
  const int ctbase = (wv >> 2) * 64;  // two column halves across 8 waves

  v8f acc[4];
#pragma unroll
  for (int j = 0; j < 4; ++j)
#pragma unroll
    for (int v = 0; v < 8; ++v) acc[j][v] = 0.0f;

#pragma unroll
  for (int kb = 0; kb < 128; kb += 32) {
    v16h a = load_frag_a(h2 + rowblk * 128, 128, rbase, kb, lane);
#pragma unroll
    for (int j = 0; j < 4; ++j) {
      v16h b = load_frag_b(lwh, 128, ctbase + j * 16, kb, lane);
      acc[j] = __builtin_amdgcn_wmma_f32_16x16x32_f16(
          false, a, false, b, (short)0, acc[j], false, false);
    }
  }

  const int mrow = rbase + ((lane >> 4) << 3);
#pragma unroll
  for (int j = 0; j < 4; ++j) {
    int col = ctbase + j * 16 + (lane & 15);
    float bb = lb[col];
#pragma unroll
    for (int v = 0; v < 8; ++v)
      em[(rowblk + mrow + v) * V_SZ + col] = acc[j][v] + bb;
  }
}

// ---------------------------------------------------------------------------
// CRF numerator (gold-path score), one thread per batch element.
// ---------------------------------------------------------------------------
__global__ void crf_score_kernel(
    const float* __restrict__ em, const int* __restrict__ x,
    const int* __restrict__ tags, const float* __restrict__ start,
    const float* __restrict__ endv, const float* __restrict__ trans,
    float* __restrict__ score) {
  int b = blockIdx.x * blockDim.x + threadIdx.x;
  if (b >= B_SZ) return;
  const int* tb = tags + (size_t)b * T_SZ;
  const int* xb = x + (size_t)b * T_SZ;
  int tg0 = tb[0];
  float sc = start[tg0] + em[((size_t)b * T_SZ) * V_SZ + tg0];
  int prev = tg0, last = tg0;
  for (int t = 1; t < T_SZ; ++t) {
    int tg = tb[t];
    if (xb[t] != 0) {
      sc += trans[prev * V_SZ + tg] + em[((size_t)b * T_SZ + t) * V_SZ + tg];
      last = tg;
    }
    prev = tg;
  }
  score[b] = sc + endv[last];
}

// ---------------------------------------------------------------------------
// CRF partition: forward algorithm. trans (64 KB) lives in LDS; 2 batch
// elements per block, 128 threads per element (thread = target tag j).
// ---------------------------------------------------------------------------
__global__ __launch_bounds__(256) void crf_partition_kernel(
    const float* __restrict__ em, const int* __restrict__ x,
    const float* __restrict__ start, const float* __restrict__ endv,
    const float* __restrict__ trans, float* __restrict__ part) {
  extern __shared__ char smem[];
  float* trs = (float*)smem;              // 128*128
  float* al  = trs + V_SZ * V_SZ;         // 2 buffers * 2 b * 128

  const int tid = threadIdx.x;
  const int bl  = tid >> 7;
  const int j   = tid & 127;
  const int b   = blockIdx.x * 2 + bl;

  for (int idx = tid; idx < V_SZ * V_SZ; idx += 256) trs[idx] = trans[idx];
  al[bl * 128 + j] = start[j] + em[((size_t)b * T_SZ) * V_SZ + j];
  __syncthreads();

  int p = 0;
  for (int t = 1; t < T_SZ; ++t) {
    const float* c = al + p * 256;
    float*       n = al + (p ^ 1) * 256;
    const bool  m   = (x[(size_t)b * T_SZ + t] != 0);
    const float emt = em[((size_t)b * T_SZ + t) * V_SZ + j];

    float mx = -1e30f;
#pragma unroll 4
    for (int i = 0; i < V_SZ; ++i)
      mx = fmaxf(mx, c[bl * 128 + i] + trs[i * V_SZ + j]);
    float s = 0.0f;
#pragma unroll 4
    for (int i = 0; i < V_SZ; ++i)
      s += __expf(c[bl * 128 + i] + trs[i * V_SZ + j] - mx);
    float nv = mx + __logf(s) + emt;
    n[bl * 128 + j] = m ? nv : c[bl * 128 + j];
    __syncthreads();
    p ^= 1;
  }

  const float* c = al + p * 256;
  float* n = al + (p ^ 1) * 256;
  n[bl * 128 + j] = c[bl * 128 + j] + endv[j];
  __syncthreads();
  if (j == 0) {
    float mx = -1e30f;
    for (int i = 0; i < V_SZ; ++i) mx = fmaxf(mx, n[bl * 128 + i]);
    float s = 0.0f;
    for (int i = 0; i < V_SZ; ++i) s += __expf(n[bl * 128 + i] - mx);
    part[b] = mx + __logf(s);
  }
}

// ---------------------------------------------------------------------------
// Final scalar: -mean(score - partition) = mean(partition - score)
// ---------------------------------------------------------------------------
__global__ void final_reduce_kernel(const float* __restrict__ score,
                                    const float* __restrict__ part,
                                    float* __restrict__ out) {
  __shared__ float red[256];
  int t = threadIdx.x;
  float s = (part[t] - score[t]) + (part[t + 256] - score[t + 256]);
  red[t] = s;
  __syncthreads();
  for (int off = 128; off > 0; off >>= 1) {
    if (t < off) red[t] += red[t + off];
    __syncthreads();
  }
  if (t == 0) out[0] = red[0] / (float)B_SZ;
}

// ---------------------------------------------------------------------------
// Host-side launch
// ---------------------------------------------------------------------------
extern "C" void kernel_launch(void* const* d_in, const int* in_sizes, int n_in,
                              void* d_out, int out_size, void* d_ws, size_t ws_size,
                              hipStream_t stream) {
  (void)in_sizes; (void)n_in; (void)out_size; (void)ws_size;

  const int*   x     = (const int*)d_in[0];
  const int*   tags  = (const int*)d_in[1];
  const float* emb   = (const float*)d_in[2];
  const float* wih0  = (const float*)d_in[3];
  const float* whh0  = (const float*)d_in[4];
  const float* bih0  = (const float*)d_in[5];
  const float* bhh0  = (const float*)d_in[6];
  const float* wih0r = (const float*)d_in[7];
  const float* whh0r = (const float*)d_in[8];
  const float* bih0r = (const float*)d_in[9];
  const float* bhh0r = (const float*)d_in[10];
  const float* wih1  = (const float*)d_in[11];
  const float* whh1  = (const float*)d_in[12];
  const float* bih1  = (const float*)d_in[13];
  const float* bhh1  = (const float*)d_in[14];
  const float* wih1r = (const float*)d_in[15];
  const float* whh1r = (const float*)d_in[16];
  const float* bih1r = (const float*)d_in[17];
  const float* bhh1r = (const float*)d_in[18];
  const float* linw  = (const float*)d_in[19];
  const float* linb  = (const float*)d_in[20];
  const float* cstart= (const float*)d_in[21];
  const float* cend  = (const float*)d_in[22];
  const float* ctrans= (const float*)d_in[23];
  float* out = (float*)d_out;

  // Workspace carve (256-byte aligned slabs)
  char* ws = (char*)d_ws;
  size_t off = 0;
  auto carve = [&](size_t bytes) -> char* {
    char* p = ws + off;
    off = (off + bytes + 255) & ~(size_t)255;
    return p;
  };
  _Float16* emb_h = (_Float16*)carve((size_t)B_SZ * T_SZ * E_SZ * 2);     // 16 MB
  _Float16* h1    = (_Float16*)carve((size_t)B_SZ * T_SZ * 128 * 2);      // 32 MB
  _Float16* h2    = (_Float16*)carve((size_t)B_SZ * T_SZ * 128 * 2);      // 32 MB
  float*    em    = (float*)   carve((size_t)B_SZ * T_SZ * V_SZ * 4);     // 64 MB
  _Float16* Wl0f  = (_Float16*)carve(256 * 128 * 2);
  _Float16* Wl0r  = (_Float16*)carve(256 * 128 * 2);
  _Float16* Wl1f  = (_Float16*)carve(256 * 192 * 2);
  _Float16* Wl1r  = (_Float16*)carve(256 * 192 * 2);
  float*    biasA = (float*)   carve(1024 * 4);
  _Float16* lwh   = (_Float16*)carve(V_SZ * V_SZ * 2);
  float*    score = (float*)   carve(B_SZ * 4);
  float*    part  = (float*)   carve(B_SZ * 4);

  // 1. pack weights -> f16 WMMA layouts
  pack_weights_kernel<<<192, 256, 0, stream>>>(
      wih0, whh0, bih0, bhh0, wih0r, whh0r, bih0r, bhh0r,
      wih1, whh1, bih1, bhh1, wih1r, whh1r, bih1r, bhh1r,
      linw, Wl0f, Wl0r, Wl1f, Wl1r, biasA, lwh);

  // 2. embedding gather -> f16
  embed_kernel<<<4096, 256, 0, stream>>>(x, emb, emb_h);

  // 3. BiLSTM layer 0 (D=64): both directions in one launch (grid.y)
  {
    const size_t smem0 = (size_t)(256 * 128 + 16 * 128) * sizeof(_Float16) + 256 * sizeof(float);
    lstm_dir_kernel<64><<<dim3(B_SZ / 16, 2), 128, smem0, stream>>>(
        emb_h, h1, Wl0f, Wl0r, biasA, biasA + 256);
  }
  // 4. BiLSTM layer 1 (D=128)
  {
    const size_t smem1 = (size_t)(256 * 192 + 16 * 192) * sizeof(_Float16) + 256 * sizeof(float);
    lstm_dir_kernel<128><<<dim3(B_SZ / 16, 2), 128, smem1, stream>>>(
        h1, h2, Wl1f, Wl1r, biasA + 512, biasA + 768);
  }

  // 5. emissions
  emissions_kernel<<<(B_SZ * T_SZ) / 64, 256, 0, stream>>>(h2, lwh, linb, em);

  // 6. CRF numerator
  crf_score_kernel<<<2, 256, 0, stream>>>(em, x, tags, cstart, cend, ctrans, score);

  // 7. CRF partition (forward algorithm), trans resident in LDS
  {
    const size_t smemp = (size_t)V_SZ * V_SZ * sizeof(float) + 512 * sizeof(float);
    crf_partition_kernel<<<B_SZ / 2, 256, smemp, stream>>>(em, x, cstart, cend, ctrans, part);
  }

  // 8. scalar NLL
  final_reduce_kernel<<<1, 256, 0, stream>>>(score, part, out);
}
